// WarpingLayer_53798760349851
// MI455X (gfx1250) — matrix operand order repaired and checked by reading, not compiled, not run on hardware
//
#include <hip/hip_runtime.h>
#include <hip/hip_bf16.h>
#include <math.h>

// ---------------------------------------------------------------------------
// Shapes (fixed by the reference)
// ---------------------------------------------------------------------------
#define B_ 16
#define T_ 4096
#define C_ 8
#define F_ 64
#define K_ 11
#define L_ 4096
#define NTILES (L_ / 16)   // 256 column tiles of 16

typedef __attribute__((ext_vector_type(16))) __bf16 v16bf;
typedef __attribute__((ext_vector_type(8)))  __bf16 v8bf;
typedef __attribute__((ext_vector_type(8)))  float  v8f;

// ---------------------------------------------------------------------------
// Kernel 1: Conv1D (SAME, K=11) + BatchNorm (inference)  -> h [B,T,F] fp32
// ---------------------------------------------------------------------------
__global__ void conv_bn_kernel(const float* __restrict__ x,
                               const float* __restrict__ w,   // [K,C,F]
                               const float* __restrict__ cb,  // [F]
                               const float* __restrict__ gamma,
                               const float* __restrict__ beta,
                               const float* __restrict__ mean,
                               const float* __restrict__ var,
                               float* __restrict__ h) {
    int idx = blockIdx.x * blockDim.x + threadIdx.x;       // [0, B*T*F)
    int f = idx & (F_ - 1);
    int t = (idx >> 6) & (T_ - 1);
    int b = idx >> 18;
    float acc = cb[f];
    #pragma unroll
    for (int k = 0; k < K_; ++k) {
        int ti = t + k - (K_ / 2);
        if (ti >= 0 && ti < T_) {
            const float* xr = x + ((size_t)b * T_ + ti) * C_;
            const float* wr = w + ((size_t)k * C_) * F_ + f;
            #pragma unroll
            for (int c = 0; c < C_; ++c)
                acc = fmaf(xr[c], wr[(size_t)c * F_], acc);
        }
    }
    float inv = rsqrtf(var[f] + 1e-3f);
    h[idx] = (acc - mean[f]) * inv * gamma[f] + beta[f];
}

// ---------------------------------------------------------------------------
// Kernel 2: softmax over time axis (per (b,f) column) + sine pos-encoding,
//           emit bf16 hb [B,T,F].  One 256-thread block per column.
// ---------------------------------------------------------------------------
__device__ __forceinline__ float waveMax(float v) {
    #pragma unroll
    for (int o = 16; o; o >>= 1) v = fmaxf(v, __shfl_xor(v, o, 32));
    return v;
}
__device__ __forceinline__ float waveSum(float v) {
    #pragma unroll
    for (int o = 16; o; o >>= 1) v += __shfl_xor(v, o, 32);
    return v;
}

__global__ void softmax_pe_kernel(const float* __restrict__ h,
                                  __bf16* __restrict__ hb) {
    int b = blockIdx.x >> 6;
    int f = blockIdx.x & (F_ - 1);
    const float* col = h + (size_t)b * T_ * F_ + f;
    __shared__ float red[8];
    int wave = threadIdx.x >> 5, lane = threadIdx.x & 31;

    // pass 1: max over T
    float m = -3.4e38f;
    for (int t = threadIdx.x; t < T_; t += 256) m = fmaxf(m, col[(size_t)t * F_]);
    m = waveMax(m);
    if (lane == 0) red[wave] = m;
    __syncthreads();
    m = waveMax(lane < 8 ? red[lane] : -3.4e38f);
    m = __shfl(m, 0, 32);
    __syncthreads();

    // pass 2: sum of exp
    float s = 0.f;
    for (int t = threadIdx.x; t < T_; t += 256) s += __expf(col[(size_t)t * F_] - m);
    s = waveSum(s);
    if (lane == 0) red[wave] = s;
    __syncthreads();
    s = waveSum(lane < 8 ? red[lane] : 0.f);
    s = __shfl(s, 0, 32);
    float invs = __builtin_amdgcn_rcpf(s);

    // sine positional encoding: timescale = (1e-4)^((f & ~1)/64)
    float tsc = __expf(-9.210340371976184f * (float)(f & ~1) * (1.0f / (float)F_));
    bool odd = (f & 1);

    for (int t = threadIdx.x; t < T_; t += 256) {
        float v = __expf(col[(size_t)t * F_] - m) * invs;
        float ang = (float)t * tsc;
        float pe = odd ? __cosf(ang) : __sinf(ang);
        hb[((size_t)b * T_ + t) * F_ + f] = (__bf16)(v + pe);
    }
}

// ---------------------------------------------------------------------------
// Kernel 3: pack dense_w [F=64, L=4096] into the per-lane WMMA B layout used here (contiguous k, same as A)
// pw[ntile][ks][lane][e]  (e = 16 contiguous-k bf16 per lane)
//   k = ks*32 + (lane>=16)*16 + e ,  n = ntile*16 + (lane&15)
// ---------------------------------------------------------------------------
__global__ void pack_w_kernel(const float* __restrict__ W,
                              __bf16* __restrict__ pw) {
    int idx = blockIdx.x * blockDim.x + threadIdx.x;  // [0, 256*2*32*16)
    int e    = idx & 15;
    int lane = (idx >> 4) & 31;
    int ks   = (idx >> 9) & 1;
    int nt   = idx >> 10;
    int k = ks * 32 + ((lane >> 4) << 4) + e;
    int n = nt * 16 + (lane & 15);
    pw[idx] = (__bf16)W[(size_t)k * L_ + n];
}

// ---------------------------------------------------------------------------
// Kernel 4 (hot): fused Dense GEMM + sigmoid + time-reduction.
//   z[b,t,l] = hb[b,t,:] @ W[:,l] ;  d = 2*sigmoid(z+bias)-1 = (1-e)/(1+e),
//   e = exp(-z);  shift[b,l] = sum_t d   (atomic accumulation)
// Block = 8 waves, one (b, ntile) per block; wave w covers t-tiles w,w+8,...
// Uniform trip count (32 iters/wave) -> scalar loop control, unrollable.
// B-operand (2 x v16bf) register-resident for the whole block lifetime.
// 64 x v_wmma_f32_16x16x32_bf16 per wave; epilogue = 1 v_exp + 1 v_rcp /elt.
// ---------------------------------------------------------------------------
__global__ void gemm_shift_kernel(const __bf16* __restrict__ hb,   // [B,T,F]
                                  const __bf16* __restrict__ pw,   // packed W
                                  const float* __restrict__ dense_b,
                                  float* __restrict__ shift) {     // [B,L]
    int blk   = blockIdx.x;            // b*NTILES + ntile
    int b     = blk >> 8;
    int ntile = blk & (NTILES - 1);
    int wave  = threadIdx.x >> 5;
    int lane  = threadIdx.x & 31;
    int lhalf = lane >> 4;             // K-half selector for A, M-half for C/D
    int lmod  = lane & 15;             // A row within tile / output column n

    // B operand: one 32B load per k-step, reused across all 32 t-tiles
    const __bf16* pwBase = pw + (size_t)ntile * (2 * 32 * 16);
    v16bf bm0 = *(const v16bf*)(pwBase + ((size_t)(0 * 32 + lane)) * 16);
    v16bf bm1 = *(const v16bf*)(pwBase + ((size_t)(1 * 32 + lane)) * 16);

    float bias = dense_b[ntile * 16 + lmod];
    float colsum = 0.0f;
    int ab = lhalf * 16;  // lane's K-chunk base within a 32-wide k-step

    // first A row for this wave; advance by 8 tiles (128 rows) per iteration
    const __bf16* arow = hb + ((size_t)b * T_ + wave * 16 + lmod) * F_;
    const size_t astep = (size_t)8 * 16 * F_;   // 8 t-tiles per step

    #pragma unroll 2
    for (int it = 0; it < (T_ / 16) / 8; ++it) {  // exactly 32 iterations
        // A operand, ks=0: k in {ab..ab+15}; ks=1: +32
        v8bf c0 = *(const v8bf*)(arow + ab);
        v8bf c1 = *(const v8bf*)(arow + ab + 8);
        v8bf c2 = *(const v8bf*)(arow + ab + 32);
        v8bf c3 = *(const v8bf*)(arow + ab + 40);
        arow += astep;
        v16bf a0, a1;
        #pragma unroll
        for (int i = 0; i < 8; ++i) {
            a0[i] = c0[i]; a0[i + 8] = c1[i];
            a1[i] = c2[i]; a1[i + 8] = c3[i];
        }
        v8f acc = {};
        acc = __builtin_amdgcn_wmma_f32_16x16x32_bf16(false, a0, false, bm0,
                                                      (short)0, acc, false, false);
        acc = __builtin_amdgcn_wmma_f32_16x16x32_bf16(false, a1, false, bm1,
                                                      (short)0, acc, false, false);
        // epilogue: 2*sigmoid(z)-1 == (1-e)/(1+e), e = exp(-z).
        // fast path: v_exp_f32 + v_rcp_f32 (no IEEE divide expansion)
        #pragma unroll
        for (int r = 0; r < 8; ++r) {
            float z = acc[r] + bias;
            float e = __expf(-z);
            float rc = __builtin_amdgcn_rcpf(1.0f + e);
            colsum = fmaf(1.0f - e, rc, colsum);
        }
    }
    // lane l and l^16 hold the two M-halves of the same column n
    colsum += __shfl_xor(colsum, 16, 32);
    if (lane < 16)
        atomicAdd(&shift[(size_t)b * L_ + ntile * 16 + lmod], colsum);
}

// ---------------------------------------------------------------------------
// Kernel 5: Gaussian shift-warp:  out[b,l,c] = sum_t M[b,l,t] * x[b,t,c]
// M = exp(-(t1 - center)^2)/AMPLITUDE underflows fp32 past |dt|~9.5, so a
// 32-wide window centered at floor(center) is numerically exact.
// One wave per (b,l); lane = one t; shuffle-reduce the 8 channels.
// ---------------------------------------------------------------------------
__global__ void warp_kernel(const float* __restrict__ x,
                            const float* __restrict__ shift,
                            float* __restrict__ out) {
    int gid  = blockIdx.x * 8 + (threadIdx.x >> 5);  // (b,l) index
    int lane = threadIdx.x & 31;
    int b = gid >> 12;
    int l = gid & (L_ - 1);

    float center = ((float)(l + 1) + shift[(size_t)b * L_ + l]);  // scale T/L = 1
    int tc = (int)floorf(center);
    int t1 = tc - 15 + lane;                                      // 1-indexed t

    float acc[C_];
    #pragma unroll
    for (int c = 0; c < C_; ++c) acc[c] = 0.0f;

    if (t1 >= 1 && t1 <= T_) {
        float dt = (float)t1 - center;
        float wgt = __expf(-dt * dt) * (1.0f / 1.772637204826652f);
        const float* xr = x + ((size_t)b * T_ + (t1 - 1)) * C_;
        #pragma unroll
        for (int c = 0; c < C_; ++c) acc[c] = wgt * xr[c];
    }
    #pragma unroll
    for (int off = 16; off; off >>= 1) {
        #pragma unroll
        for (int c = 0; c < C_; ++c) acc[c] += __shfl_down(acc[c], off, 32);
    }
    if (lane == 0) {
        float* o = out + ((size_t)b * L_ + l) * C_;
        #pragma unroll
        for (int c = 0; c < C_; ++c) o[c] = acc[c];
    }
}

// ---------------------------------------------------------------------------
// Launcher. Workspace layout (all offsets 256B-aligned):
//   [0,16MB)        h   fp32 [B,T,F]
//   [16MB,24MB)     hb  bf16 [B,T,F]
//   [24MB,+512KB)   pw  bf16 packed W
//   [24.5MB,+256KB) shift fp32 [B,L]
// ---------------------------------------------------------------------------
extern "C" void kernel_launch(void* const* d_in, const int* in_sizes, int n_in,
                              void* d_out, int out_size, void* d_ws, size_t ws_size,
                              hipStream_t stream) {
    const float* x       = (const float*)d_in[0];
    const float* conv_w  = (const float*)d_in[1];
    const float* conv_b  = (const float*)d_in[2];
    const float* gamma   = (const float*)d_in[3];
    const float* beta    = (const float*)d_in[4];
    const float* mean    = (const float*)d_in[5];
    const float* var     = (const float*)d_in[6];
    const float* dense_w = (const float*)d_in[7];
    const float* dense_b = (const float*)d_in[8];
    float* out = (float*)d_out;

    char* ws = (char*)d_ws;
    float*  h     = (float*)(ws);
    __bf16* hb    = (__bf16*)(ws + (size_t)16 * 1024 * 1024);
    __bf16* pw    = (__bf16*)(ws + (size_t)24 * 1024 * 1024);
    float*  shift = (float*)(ws + (size_t)24 * 1024 * 1024 + 512 * 1024);

    // 1. Conv1D + BN
    conv_bn_kernel<<<(B_ * T_ * F_) / 256, 256, 0, stream>>>(
        x, conv_w, conv_b, gamma, beta, mean, var, h);

    // 2. Softmax over T + positional encoding -> bf16
    softmax_pe_kernel<<<B_ * F_, 256, 0, stream>>>(h, hb);

    // 3. Pack dense_w into WMMA B layout
    pack_w_kernel<<<(NTILES * 2 * 32 * 16) / 256, 256, 0, stream>>>(dense_w, pw);

    // 4. Zero the shift accumulator (graph-capture-safe async memset)
    hipMemsetAsync(shift, 0, (size_t)B_ * L_ * sizeof(float), stream);

    // 5. Fused WMMA GEMM + sigmoid + time reduction
    gemm_shift_kernel<<<B_ * NTILES, 256, 0, stream>>>(hb, pw, dense_b, shift);

    // 6. Windowed Gaussian warp matmul -> out [B,L,C]
    warp_kernel<<<(B_ * L_) / 8, 256, 0, stream>>>(x, shift, out);
}
